// CrossAttentionBlock3D_86930138071202
// MI455X (gfx1250) — matrix operand order repaired and checked by reading, not compiled, not run on hardware
//
#include <hip/hip_runtime.h>

// ---------------------------------------------------------------------------
// CrossAttentionBlock3D for MI455X (gfx1250): bf16 WMMA everywhere,
// TDM tensor_load_to_lds + global_load_async_to_lds_b128 for attention tiles.
// B=2, C=512, D=H=W=32 (S=32768), L=256, CTX=768, NH=8, HD=64, G=8
// ---------------------------------------------------------------------------

#define BB   2
#define CC   512
#define SS   32768
#define LL   256
#define CTXD 768
#define NHH  8
#define HDD  64
#define GG   8
#define EPSF 1e-5f

typedef __attribute__((ext_vector_type(16))) __bf16 v16bf;
typedef __attribute__((ext_vector_type(8)))  __bf16 v8bf;
typedef __attribute__((ext_vector_type(8)))  float  v8f;
typedef __attribute__((ext_vector_type(4)))  unsigned int v4u;
typedef __attribute__((ext_vector_type(8)))  int v8i;
typedef __attribute__((ext_vector_type(4)))  int v4i;

// ---------------------------------------------------------------------------
// WMMA helper: D = A(16x32 bf16) * B(32x16 bf16) + C (f32)
// ---------------------------------------------------------------------------
__device__ __forceinline__ v8f wmma_bf16(v16bf a, v16bf b, v8f c) {
  return __builtin_amdgcn_wmma_f32_16x16x32_bf16(false, a, false, b,
                                                 (short)0, c, false, false);
}

// A-fragment (16x32, M x K), row-major source: lane&15 = row M,
// two contiguous 8-elem K chunks at koff and koff+16 (koff = (lane>>4)*8).
__device__ __forceinline__ v16bf load_afrag(const __bf16* base, int row_stride, int lane) {
  int m = lane & 15;
  int koff = (lane >> 4) << 3;
  const __bf16* p = base + (size_t)m * row_stride + koff;
  v8bf lo = *(const v8bf*)(p);
  v8bf hi = *(const v8bf*)(p + 16);
  v16bf a;
#pragma unroll
  for (int i = 0; i < 8; ++i) { a[i] = lo[i]; a[i + 8] = hi[i]; }
  return a;
}

__device__ __forceinline__ v16bf load_afrag_f32(const float* base, int row_stride, int lane) {
  int m = lane & 15;
  int koff = (lane >> 4) << 3;
  const float* p = base + (size_t)m * row_stride + koff;
  v16bf a;
#pragma unroll
  for (int i = 0; i < 8; ++i) {
    a[i]     = (__bf16)p[i];
    a[i + 8] = (__bf16)p[i + 16];
  }
  return a;
}

// B-fragment (32x16, K x N), source stored TRANSPOSED [n][k]:
// lane&15 = column N, two contiguous 8-elem K chunks at koff and koff+16 (koff = (lane>>4)*8).
__device__ __forceinline__ v16bf load_bfrag(const __bf16* baseT, int row_stride, int lane) {
  int n = lane & 15;
  int koff = (lane >> 4) << 3;
  const __bf16* p = baseT + (size_t)n * row_stride + koff;
  v8bf lo = *(const v8bf*)(p);
  v8bf hi = *(const v8bf*)(p + 16);
  v16bf b;
#pragma unroll
  for (int i = 0; i < 8; ++i) { b[i] = lo[i]; b[i + 8] = hi[i]; }
  return b;
}

__device__ __forceinline__ v16bf load_bfrag_f32(const float* baseT, int row_stride, int lane) {
  int n = lane & 15;
  int koff = (lane >> 4) << 3;
  const float* p = baseT + (size_t)n * row_stride + koff;
  v16bf b;
#pragma unroll
  for (int i = 0; i < 8; ++i) { b[i] = (__bf16)p[i]; b[i + 8] = (__bf16)p[i + 16]; }
  return b;
}

__device__ __forceinline__ float wave_sum(float v) {
#pragma unroll
  for (int m = 16; m; m >>= 1) v += __shfl_xor(v, m, 32);
  return v;
}

// ---------------------------------------------------------------------------
// 1) GroupNorm statistics: partial sums per (b,g) via atomics
// ---------------------------------------------------------------------------
__global__ void gn_stats_kernel(const float* __restrict__ x, float* __restrict__ raw) {
  int bg = blockIdx.x;          // 0..15  (b*8+g)
  int chunk = blockIdx.y;       // 0..31  (2 channels each)
  int b = bg >> 3, g = bg & 7;
  int c0 = g * 64 + chunk * 2;
  const float4* p = (const float4*)(x + ((size_t)(b * CC + c0)) * SS);
  float s = 0.f, q = 0.f;
  for (int i = threadIdx.x; i < (2 * SS) / 4; i += 256) {
    float4 v = p[i];
    s += v.x + v.y + v.z + v.w;
    q += v.x * v.x + v.y * v.y + v.z * v.z + v.w * v.w;
  }
  s = wave_sum(s); q = wave_sum(q);
  __shared__ float rs[8], rq[8];
  int lane = threadIdx.x & 31, wv = threadIdx.x >> 5;
  if (lane == 0) { rs[wv] = s; rq[wv] = q; }
  __syncthreads();
  if (threadIdx.x == 0) {
    float S = 0.f, Q = 0.f;
    for (int i = 0; i < 8; ++i) { S += rs[i]; Q += rq[i]; }
    atomicAdd(&raw[bg * 2 + 0], S);
    atomicAdd(&raw[bg * 2 + 1], Q);
  }
}

__global__ void gn_finalize_kernel(const float* __restrict__ raw, float* __restrict__ fin) {
  int i = threadIdx.x;
  if (i < 16) {
    const float inv = 1.f / (64.f * (float)SS);
    float mu = raw[2 * i] * inv;
    float var = raw[2 * i + 1] * inv - mu * mu;
    fin[2 * i] = mu;
    fin[2 * i + 1] = rsqrtf(var + EPSF);
  }
}

// ---------------------------------------------------------------------------
// 2) LayerNorm(context) -> bf16, row-major (512 rows x 768)
// ---------------------------------------------------------------------------
__global__ void ln_kernel(const float* __restrict__ ctx, const float* __restrict__ w,
                          const float* __restrict__ bia, __bf16* __restrict__ ctxn) {
  int row = blockIdx.x;  // 0..511
  const float* p = ctx + (size_t)row * CTXD;
  float s = 0.f, q = 0.f;
  for (int i = threadIdx.x; i < CTXD; i += 256) { float v = p[i]; s += v; q += v * v; }
  s = wave_sum(s); q = wave_sum(q);
  __shared__ float red[18];
  int lane = threadIdx.x & 31, wv = threadIdx.x >> 5;
  if (lane == 0) { red[wv] = s; red[8 + wv] = q; }
  __syncthreads();
  if (threadIdx.x == 0) {
    float S = 0.f, Q = 0.f;
    for (int i = 0; i < 8; ++i) { S += red[i]; Q += red[8 + i]; }
    float mu = S * (1.f / (float)CTXD);
    float var = Q * (1.f / (float)CTXD) - mu * mu;
    red[16] = mu;
    red[17] = rsqrtf(var + EPSF);
  }
  __syncthreads();
  float mu = red[16], rstd = red[17];
  for (int i = threadIdx.x; i < CTXD; i += 256) {
    float v = (p[i] - mu) * rstd * w[i] + bia[i];
    ctxn[(size_t)row * CTXD + i] = (__bf16)v;
  }
}

// ---------------------------------------------------------------------------
// 3) K/V GEMM: (512 ctx rows x 768) x (768 x 512).  scale folded into K.
//    kbuf stored (b,h,l,e)  [B-layout for QK^T GEMM]
//    vT   stored (b,h,e,l)  [B-layout for P*V GEMM]
// ---------------------------------------------------------------------------
__global__ void kv_kernel(const __bf16* __restrict__ ctxn,
                          const float* __restrict__ kw, const float* __restrict__ kb,
                          const float* __restrict__ vw, const float* __restrict__ vb,
                          __bf16* __restrict__ kbuf, __bf16* __restrict__ vT) {
  int m0 = blockIdx.x * 16;            // ctx row tile
  int lane = threadIdx.x & 31, wv = threadIdx.x >> 5;   // 8 waves
  int n0 = wv * 64;
  v8f accK[4] = {}, accV[4] = {};
  for (int k0 = 0; k0 < CTXD; k0 += 32) {
    v16bf a = load_afrag(ctxn + (size_t)m0 * CTXD + k0, CTXD, lane);
#pragma unroll
    for (int t = 0; t < 4; ++t) {
      v16bf bk = load_bfrag_f32(kw + (size_t)(n0 + t * 16) * CTXD + k0, CTXD, lane);
      accK[t] = wmma_bf16(a, bk, accK[t]);
      v16bf bv = load_bfrag_f32(vw + (size_t)(n0 + t * 16) * CTXD + k0, CTXD, lane);
      accV[t] = wmma_bf16(a, bv, accV[t]);
    }
  }
  int nlo = lane & 15, half = lane >> 4;
#pragma unroll
  for (int t = 0; t < 4; ++t) {
    int o = n0 + t * 16 + nlo;
    int h = o >> 6, e = o & 63;
    float kbias = kb[o], vbias = vb[o];
#pragma unroll
    for (int r = 0; r < 8; ++r) {
      int m = m0 + r + 8 * half;       // global ctx row = b*256 + l
      int b = m >> 8, l = m & 255;
      size_t bh = (size_t)(b * NHH + h);
      kbuf[(bh * LL + l) * HDD + e] = (__bf16)((accK[t][r] + kbias) * 0.125f);
      vT[(bh * HDD + e) * LL + l]   = (__bf16)(accV[t][r] + vbias);
    }
  }
}

// ---------------------------------------------------------------------------
// 4) Q GEMM (GroupNorm fused into B-tile stage):
//    q_w(512x512) x h(512 x 32768 per batch)  -> q_t stored (b, s, o) bf16
// ---------------------------------------------------------------------------
__global__ void q_gemm_kernel(const float* __restrict__ x, const float* __restrict__ qw,
                              const float* __restrict__ qb, const float* __restrict__ stats,
                              const float* __restrict__ gnw, const float* __restrict__ gnb,
                              __bf16* __restrict__ qt) {
  __shared__ __bf16 As[64 * 32];    // [m][k]
  __shared__ __bf16 Bs[128 * 32];   // [s][c] (transposed for B-frags)
  int b  = blockIdx.z;
  int m0 = blockIdx.y * 64;
  int s0 = blockIdx.x * 128;
  int tid = threadIdx.x, lane = tid & 31, wv = tid >> 5;
  int wm = wv & 3, wn = wv >> 2;
  v8f acc[4] = {};
  for (int k0 = 0; k0 < CC; k0 += 32) {
    __syncthreads();
    // stage A: 64x32 f32 -> bf16
    for (int f = tid; f < 512; f += 256) {
      int row = f >> 3, kk = (f & 7) * 4;
      float4 v = *(const float4*)(qw + (size_t)(m0 + row) * CC + k0 + kk);
      As[row * 32 + kk + 0] = (__bf16)v.x;
      As[row * 32 + kk + 1] = (__bf16)v.y;
      As[row * 32 + kk + 2] = (__bf16)v.z;
      As[row * 32 + kk + 3] = (__bf16)v.w;
    }
    // stage B: x tile 32c x 128s, normalize (GroupNorm) on the fly, transpose
    for (int f = tid; f < 1024; f += 256) {
      int ci = f >> 5, si = (f & 31) * 4;
      int cc = k0 + ci;
      int g  = cc >> 6;
      float mu   = stats[(b * GG + g) * 2 + 0];
      float rstd = stats[(b * GG + g) * 2 + 1];
      float wsc = gnw[cc] * rstd;
      float bof = gnb[cc] - mu * wsc;
      const float* src = x + ((size_t)(b * CC + cc)) * SS + s0 + si;
      if (k0 + 32 < CC)
        __builtin_prefetch(src + (size_t)32 * SS, 0, 1);   // global_prefetch_b8
      float4 v = *(const float4*)src;
      Bs[(si + 0) * 32 + ci] = (__bf16)(v.x * wsc + bof);
      Bs[(si + 1) * 32 + ci] = (__bf16)(v.y * wsc + bof);
      Bs[(si + 2) * 32 + ci] = (__bf16)(v.z * wsc + bof);
      Bs[(si + 3) * 32 + ci] = (__bf16)(v.w * wsc + bof);
    }
    __syncthreads();
    v16bf a = load_afrag(As + (wm * 16) * 32, 32, lane);
#pragma unroll
    for (int t = 0; t < 4; ++t) {
      v16bf bf = load_bfrag(Bs + (wn * 64 + t * 16) * 32, 32, lane);
      acc[t] = wmma_bf16(a, bf, acc[t]);
    }
  }
  int nlo = lane & 15, half = lane >> 4;
#pragma unroll
  for (int t = 0; t < 4; ++t) {
    int s = s0 + wn * 64 + t * 16 + nlo;
    int obase = m0 + wm * 16 + 8 * half;
    __bf16* dst = qt + ((size_t)(b * SS + s)) * CC + obase;
#pragma unroll
    for (int r = 0; r < 8; ++r) dst[r] = (__bf16)(acc[t][r] + qb[obase + r]);
  }
}

// ---------------------------------------------------------------------------
// 5) Attention: per (b,h,64-row s-tile). K tile via TDM tensor_load_to_lds,
//    V^T tile via global_load_async_to_lds_b128.  LDS 64KB total;
//    P overwrites the K tile after QK^T + softmax. out stored (b, s, o) bf16.
// ---------------------------------------------------------------------------
__global__ void attn_kernel(const __bf16* __restrict__ qt, const __bf16* __restrict__ kbuf,
                            const __bf16* __restrict__ vT, __bf16* __restrict__ outT) {
  __shared__ __bf16 ldsA[LL * HDD];   // K tile [l][e], later P tile [s][l] (32KB)
  __shared__ __bf16 ldsB[HDD * LL];   // V^T tile [e][l] (32KB)
  int s0 = blockIdx.x * 64;
  int h  = blockIdx.y;
  int b  = blockIdx.z;
  int tid = threadIdx.x, lane = tid & 31, wv = tid >> 5;   // 4 waves
  size_t bh = (size_t)(b * NHH + h);

  // --- K tile: Tensor Data Mover.  D# per ISA 08_async_tensor §8:
  // 2D tensor, data_size=2B, dim0=64 (e, contiguous), dim1=256 (l),
  // tile = whole tensor, dim0_stride = 64.  Issued by wave 0 only
  // (TDM ignores EXEC; one DMA per workgroup).
  if (wv == 0) {
    unsigned lds_off = (unsigned)(uintptr_t)&ldsA[0];   // low 32 bits = LDS addr
    unsigned long long ga = (unsigned long long)(uintptr_t)(kbuf + bh * LL * HDD);
    v4u g0;
    g0[0] = 1u;                                          // count=1, user mode
    g0[1] = lds_off;                                     // lds_addr
    g0[2] = (unsigned)(ga & 0xffffffffu);                // global_addr[31:0]
    g0[3] = (unsigned)((ga >> 32) & 0x01ffffffu) | 0x80000000u; // [56:32] | type=2
    v8i g1;
    g1[0] = 0x00010000;          // workgroup_mask=0, data_size=1 (2 bytes)
    g1[1] = (int)(64u << 16);    // tensor_dim0 = 64  (bits [79:48], low half)
    g1[2] = (int)(256u << 16);   // tensor_dim1 = 256 (bits [111:80], low half)
    g1[3] = (int)(64u << 16);    // tile_dim0 = 64    (bits [127:112])
    g1[4] = 256;                 // tile_dim1 = 256   (bits [143:128])
    g1[5] = 64;                  // tensor_dim0_stride = 64 (bits [207:160])
    g1[6] = 0;
    g1[7] = 0;
    v4i z4 = {};
    v8i z8 = {};
    __builtin_amdgcn_tensor_load_to_lds(g0, g1, z4, z4, z8, 0);
  }

  // --- V^T tile: per-lane async copy global -> LDS (ASYNCcnt path).
  const uint4* vsrc = (const uint4*)(vT + bh * HDD * LL);
  for (int i = tid; i < 2048; i += 128) {
    unsigned ldsa = (unsigned)(uintptr_t)&((uint4*)ldsB)[i];
    const uint4* g = vsrc + i;
    asm volatile("global_load_async_to_lds_b128 %0, %1, off"
                 :: "v"(ldsa), "v"(g) : "memory");
  }
  asm volatile("s_wait_asynccnt 0x0" ::: "memory");
  __builtin_amdgcn_s_wait_tensorcnt(0);
  __syncthreads();

  // QK^T: M = 16 s-rows per wave, N = 256 (16 tiles), K = 64
  v8f acc[16] = {};
  const __bf16* qrow = qt + ((size_t)(b * SS + s0 + wv * 16)) * CC + h * HDD;
#pragma unroll
  for (int ke = 0; ke < 2; ++ke) {
    v16bf a = load_afrag(qrow + ke * 32, CC, lane);
#pragma unroll
    for (int t = 0; t < 16; ++t) {
      v16bf bf = load_bfrag(ldsA + (t * 16) * HDD + ke * 32, HDD, lane);
      acc[t] = wmma_bf16(a, bf, acc[t]);
    }
  }

  // softmax over l (scale already folded into K): per-row reduce across
  // the 16-lane half + the 16 register tiles.
#pragma unroll
  for (int r = 0; r < 8; ++r) {
    float mx = -1e30f;
#pragma unroll
    for (int t = 0; t < 16; ++t) mx = fmaxf(mx, acc[t][r]);
#pragma unroll
    for (int off = 8; off; off >>= 1) mx = fmaxf(mx, __shfl_xor(mx, off, 32));
    float sum = 0.f;
#pragma unroll
    for (int t = 0; t < 16; ++t) { float e = __expf(acc[t][r] - mx); acc[t][r] = e; sum += e; }
#pragma unroll
    for (int off = 8; off; off >>= 1) sum += __shfl_xor(sum, off, 32);
    float rc = 1.f / sum;
#pragma unroll
    for (int t = 0; t < 16; ++t) acc[t][r] *= rc;
  }

  __syncthreads();   // all waves done reading K tile
  int nlo = lane & 15, half = lane >> 4;
#pragma unroll
  for (int t = 0; t < 16; ++t)
#pragma unroll
    for (int r = 0; r < 8; ++r)
      ldsA[(wv * 16 + r + 8 * half) * LL + t * 16 + nlo] = (__bf16)acc[t][r];
  __syncthreads();

  // P * V: M = 16 s-rows, N = 64 (4 tiles), K = 256
  v8f oac[4] = {};
  for (int k0 = 0; k0 < LL; k0 += 32) {
    v16bf a = load_afrag(ldsA + (wv * 16) * LL + k0, LL, lane);
#pragma unroll
    for (int t = 0; t < 4; ++t) {
      v16bf bf = load_bfrag(ldsB + (t * 16) * LL + k0, LL, lane);
      oac[t] = wmma_bf16(a, bf, oac[t]);
    }
  }
#pragma unroll
  for (int t = 0; t < 4; ++t)
#pragma unroll
    for (int r = 0; r < 8; ++r) {
      int s = s0 + wv * 16 + r + 8 * half;
      outT[((size_t)(b * SS + s)) * CC + h * HDD + t * 16 + nlo] = (__bf16)oac[t][r];
    }
}

// ---------------------------------------------------------------------------
// 6) Proj GEMM + bias + residual, fp32 out (b, o, s)
// ---------------------------------------------------------------------------
__global__ void proj_kernel(const __bf16* __restrict__ outT, const float* __restrict__ pw,
                            const float* __restrict__ pb, const float* __restrict__ x,
                            float* __restrict__ out) {
  int b  = blockIdx.z;
  int m0 = blockIdx.y * 64;
  int s0 = blockIdx.x * 128;
  int tid = threadIdx.x, lane = tid & 31, wv = tid >> 5;
  int wm = wv & 3, wn = wv >> 2;
  v8f acc[4] = {};
  const float* arow = pw + (size_t)(m0 + wm * 16) * CC;
  for (int k0 = 0; k0 < CC; k0 += 32) {
    v16bf a = load_afrag_f32(arow + k0, CC, lane);
#pragma unroll
    for (int t = 0; t < 4; ++t) {
      const __bf16* bT = outT + ((size_t)(b * SS + s0 + wn * 64 + t * 16)) * CC + k0;
      v16bf bf = load_bfrag(bT, CC, lane);
      acc[t] = wmma_bf16(a, bf, acc[t]);
    }
  }
  int nlo = lane & 15, half = lane >> 4;
#pragma unroll
  for (int t = 0; t < 4; ++t) {
    int s = s0 + wn * 64 + t * 16 + nlo;
#pragma unroll
    for (int r = 0; r < 8; ++r) {
      int o = m0 + wm * 16 + 8 * half + r;
      size_t idx = ((size_t)(b * CC + o)) * SS + s;
      out[idx] = acc[t][r] + pb[o] + x[idx];
    }
  }
}

// ---------------------------------------------------------------------------
// launch
// ---------------------------------------------------------------------------
extern "C" void kernel_launch(void* const* d_in, const int* in_sizes, int n_in,
                              void* d_out, int out_size, void* d_ws, size_t ws_size,
                              hipStream_t stream) {
  const float* x      = (const float*)d_in[0];
  const float* ctx    = (const float*)d_in[1];
  const float* gn_w   = (const float*)d_in[2];
  const float* gn_b   = (const float*)d_in[3];
  const float* ln_w   = (const float*)d_in[4];
  const float* ln_b   = (const float*)d_in[5];
  const float* q_w    = (const float*)d_in[6];
  const float* q_b    = (const float*)d_in[7];
  const float* k_w    = (const float*)d_in[8];
  const float* k_b    = (const float*)d_in[9];
  const float* v_w    = (const float*)d_in[10];
  const float* v_b    = (const float*)d_in[11];
  const float* proj_w = (const float*)d_in[12];
  const float* proj_b = (const float*)d_in[13];
  float* out = (float*)d_out;

  char* ws = (char*)d_ws;
  float*  stats_raw = (float*)(ws + 0);                 // 128 B
  float*  stats_fin = (float*)(ws + 256);               // 128 B
  __bf16* ctxn = (__bf16*)(ws + 512);                   // 786432 B
  __bf16* kbuf = (__bf16*)(ws + 512 + 786432);          // 524288 B
  __bf16* vTb  = (__bf16*)(ws + 512 + 786432 + 524288); // 524288 B
  size_t base  = 512 + 786432 + 2 * 524288;             // 1835520
  __bf16* qt   = (__bf16*)(ws + base);                  // 64 MB
  __bf16* outT = (__bf16*)(ws + base + (size_t)BB * SS * CC * 2);

  (void)hipMemsetAsync(stats_raw, 0, 256, stream);

  gn_stats_kernel<<<dim3(16, 32), 256, 0, stream>>>(x, stats_raw);
  gn_finalize_kernel<<<1, 32, 0, stream>>>(stats_raw, stats_fin);
  ln_kernel<<<BB * LL, 256, 0, stream>>>(ctx, ln_w, ln_b, ctxn);
  kv_kernel<<<32, 256, 0, stream>>>(ctxn, k_w, k_b, v_w, v_b, kbuf, vTb);
  q_gemm_kernel<<<dim3(SS / 128, CC / 64, BB), 256, 0, stream>>>(
      x, q_w, q_b, stats_fin, gn_w, gn_b, qt);
  attn_kernel<<<dim3(SS / 64, NHH, BB), 128, 0, stream>>>(qt, kbuf, vTb, outT);
  proj_kernel<<<dim3(SS / 128, CC / 64, BB), 256, 0, stream>>>(
      outT, proj_w, proj_b, x, out);
}
